// ODEDecoder_14714557956414
// MI455X (gfx1250) — matrix operand order repaired and checked by reading, not compiled, not run on hardware
//
#include <hip/hip_runtime.h>
#include <hip/hip_bf16.h>

// MI455X / gfx1250, wave32. WMMA f16->f32 16x16x32; TDM latent staging; v_tanh_f32 epilogues.

typedef __attribute__((ext_vector_type(16))) _Float16 v16h;
typedef __attribute__((ext_vector_type(8)))  float    v8f;
typedef unsigned int v4u __attribute__((ext_vector_type(4)));
typedef int          v8i __attribute__((ext_vector_type(8)));
typedef int          v4i __attribute__((ext_vector_type(4)));

union AB32 { v16h h; uint4 q[2]; };

__device__ __forceinline__ float fast_tanh(float x) {
#if __has_builtin(__builtin_amdgcn_tanhf)
  return __builtin_amdgcn_tanhf(x);          // v_tanh_f32 (TRANS op, co-executes with WMMA)
#else
  float y;
  asm("v_tanh_f32 %0, %1" : "=v"(y) : "v"(x));
  return y;
#endif
}

// One wave computes a 16x16 f32 tile: C += A(16xK, LDS f16 row-major, stride lda)
//                                        * B(K x 16 tile; Bt = transposed weights [n][k], stride ldb)
// A lane layout (16-bit A 16x32):  m = lane&15, chunks [kg*8,+8) and [kg*8+16,+8), kg = lane>>4
// B lane layout (16-bit B 32x16):  n = lane&15, chunks [kg*8,+8) and [kg*8+16,+8), kg = lane>>4
__device__ __forceinline__ v8f wmma_tile(const _Float16* __restrict__ As, int lda,
                                         const _Float16* __restrict__ Bt, int ldb,
                                         int K, v8f c, int lane) {
  const int mr = lane & 15;
  const int kg = lane >> 4;
  const _Float16* arow = As + mr * lda + kg * 8;
  const _Float16* brow = Bt + mr * ldb + kg * 8;
  for (int k0 = 0; k0 < K; k0 += 32) {
    AB32 a, b;
    a.q[0] = *(const uint4*)(arow + k0);
    a.q[1] = *(const uint4*)(arow + k0 + 16);
    b.q[0] = *(const uint4*)(brow + k0);
    b.q[1] = *(const uint4*)(brow + k0 + 16);
    c = __builtin_amdgcn_wmma_f32_16x16x32_f16(false, a.h, false, b.h,
                                               (short)0, c, false, false);
  }
  return c;
}

// ---------------- weight packing (f32 row-major [K][N] -> f16 transposed [N][K]) -------------
__global__ void pack_transpose_f16(const float* __restrict__ W, _Float16* __restrict__ Wt,
                                   int K, int N) {
  int i = blockIdx.x * 256 + threadIdx.x;
  if (i >= K * N) return;
  int n = i % N, k = i / N;
  Wt[n * K + k] = (_Float16)W[k * N + n];
}

// Wh is 257x256 with original inp order [x | latent(128) | z_(128)].
// Decoder K layout is reordered to [latent(0..127) | x(128) | z_(129..256) | pad(257..287)]
// so the TDM can drop latent rows at 4B-aligned LDS row starts.
__global__ void pack_wh_f16(const float* __restrict__ Wh, _Float16* __restrict__ Wht) {
  int i = blockIdx.x * 256 + threadIdx.x;
  if (i >= 256 * 288) return;
  int n = i / 288, k = i % 288;
  float v;
  if (k < 128)        v = Wh[(1 + k) * 256 + n];   // latent rows 1..128
  else if (k == 128)  v = Wh[0 * 256 + n];         // x row 0
  else if (k <= 256)  v = Wh[k * 256 + n];         // z_ rows 129..256
  else                v = 0.0f;
  Wht[i] = (_Float16)v;
}

__global__ void pack_time_row(const float* __restrict__ W1, float* __restrict__ w1time) {
  int n = threadIdx.x;            // 256 threads
  w1time[n] = W1[256 * 256 + n];  // W1 row 256 = time weights
}

// ---------------- ODE: fixed-step RK4 over the shared grid -----------------
// 16 blocks x 256 threads (8 waves). Block = 16 batch rows; weights end up register-resident.
#define NSUB 2

__global__ __launch_bounds__(256) void ode_rk4_kernel(
    const float* __restrict__ z,
    const float* __restrict__ b1, const float* __restrict__ b2, const float* __restrict__ b3,
    const _Float16* __restrict__ W1t, const float* __restrict__ w1time,
    const _Float16* __restrict__ W2t, const _Float16* __restrict__ W3t,
    _Float16* __restrict__ latent)   // (B, P, 128) f16, latent[b][p] = L(t_{p+1})
{
  __shared__ __align__(16) float    vL[16 * 128];
  __shared__ __align__(16) float    kacc[16 * 128];
  __shared__ __align__(16) float    ktmp[16 * 128];
  __shared__ __align__(16) _Float16 abuf[16 * 256];   // vf input [L | z] in f16
  __shared__ __align__(16) _Float16 h1b[16 * 256];
  __shared__ __align__(16) _Float16 h2b[16 * 256];

  const int tid  = threadIdx.x;
  const int wave = tid >> 5;
  const int lane = tid & 31;
  const int blk  = blockIdx.x;

  // Pull weight panels toward L2 (global_prefetch_b8); reused ~2k times per block.
  for (int off = tid * 64; off < 256 * 256; off += 256 * 64) {
    __builtin_prefetch(W1t + off, 0, 1);
    __builtin_prefetch(W2t + off, 0, 1);
  }
  for (int off = tid * 64; off < 128 * 256; off += 256 * 64)
    __builtin_prefetch(W3t + off, 0, 1);

  // init state: vL = z[:, :128]; frozen z-part of A matrix (columns 128..255)
  for (int idx = tid; idx < 16 * 128; idx += 256) {
    int r = idx >> 7, cc = idx & 127;
    int b = blk * 16 + r;
    vL[idx] = z[b * 256 + cc];
    abuf[r * 256 + 128 + cc] = (_Float16)z[b * 256 + 128 + cc];
  }
  __syncthreads();

  const float dt = (1.0f / 256.0f) / (float)NSUB;

  for (int p = 0; p < 256; ++p) {
    float t0 = (float)p * (1.0f / 256.0f);
    for (int s = 0; s < NSUB; ++s) {
      float tb = t0 + (float)s * dt;
      #pragma unroll
      for (int st = 0; st < 4; ++st) {
        const float ca = (st == 0) ? 0.0f : (st == 3 ? dt : 0.5f * dt);
        const float ts = tb + ca;
        const float kw = (st == 0 || st == 3) ? 1.0f : 2.0f;

        // stage input: A L-part = f16(vL + ca * k_prev)
        for (int idx = tid; idx < 16 * 128; idx += 256) {
          int r = idx >> 7, cc = idx & 127;
          float vv = vL[idx];
          if (st > 0) vv += ca * ktmp[idx];
          abuf[r * 256 + cc] = (_Float16)vv;
        }
        __syncthreads();

        // ---- h1 = tanh(vt@W1 + b1 + t*w1time) ----
        {
          v8f c0 = {}, c1 = {};
          c0 = wmma_tile(abuf, 256, W1t + (size_t)(wave * 32) * 256, 256, 256, c0, lane);
          c1 = wmma_tile(abuf, 256, W1t + (size_t)(wave * 32 + 16) * 256, 256, 256, c1, lane);
          int nA = wave * 32 + (lane & 15), nB = nA + 16;
          float bA = b1[nA] + ts * w1time[nA];
          float bB = b1[nB] + ts * w1time[nB];
          int hi = lane >> 4;
          #pragma unroll
          for (int i = 0; i < 8; ++i) {
            h1b[(i + hi * 8) * 256 + nA] = (_Float16)fast_tanh(c0[i] + bA);
            h1b[(i + hi * 8) * 256 + nB] = (_Float16)fast_tanh(c1[i] + bB);
          }
        }
        __syncthreads();

        // ---- h2 = tanh(h1@W2 + b2) ----
        {
          v8f d0 = {}, d1 = {};
          d0 = wmma_tile(h1b, 256, W2t + (size_t)(wave * 32) * 256, 256, 256, d0, lane);
          d1 = wmma_tile(h1b, 256, W2t + (size_t)(wave * 32 + 16) * 256, 256, 256, d1, lane);
          int nA = wave * 32 + (lane & 15), nB = nA + 16;
          float bA = b2[nA], bB = b2[nB];
          int hi = lane >> 4;
          #pragma unroll
          for (int i = 0; i < 8; ++i) {
            h2b[(i + hi * 8) * 256 + nA] = (_Float16)fast_tanh(d0[i] + bA);
            h2b[(i + hi * 8) * 256 + nB] = (_Float16)fast_tanh(d1[i] + bB);
          }
        }
        __syncthreads();

        // ---- dL = h2@W3 + b3  (N=128: one tile per wave) ----
        {
          v8f e0 = {};
          e0 = wmma_tile(h2b, 256, W3t + (size_t)(wave * 16) * 256, 256, 256, e0, lane);
          int n = wave * 16 + (lane & 15);
          float bb = b3[n];
          int hi = lane >> 4;
          #pragma unroll
          for (int i = 0; i < 8; ++i)
            ktmp[(i + hi * 8) * 128 + n] = e0[i] + bb;
        }
        __syncthreads();

        // RK4 accumulate
        for (int idx = tid; idx < 16 * 128; idx += 256) {
          float kv = ktmp[idx];
          kacc[idx] = (st == 0) ? kv : kacc[idx] + kw * kv;
        }
      }
      // vL += dt/6 * (k1 + 2k2 + 2k3 + k4)
      for (int idx = tid; idx < 16 * 128; idx += 256)
        vL[idx] += (dt * (1.0f / 6.0f)) * kacc[idx];
      __syncthreads();
    }
    // store L(t_{p+1})
    for (int idx = tid; idx < 16 * 128; idx += 256) {
      int r = idx >> 7, cc = idx & 127;
      int b = blk * 16 + r;
      latent[((size_t)b * 256 + (size_t)p) * 128 + cc] = (_Float16)vL[idx];
    }
  }
}

// ---------------- decoder: hidden = relu([latent|x|z_]@Wh+bh); mu/sigma GEMVs --------------
// 4096 blocks x 256 threads; one 16-row M-tile of the 65536x256x288 GEMM.
// Latent tile (16x128 f16, contiguous 4KB in global) staged into LDS by the Tensor Data Mover
// with LDS padding (64 dwords data + 80 dwords pad = 576B row stride), overlapped with the
// x/z column fill; published via s_wait_tensorcnt + barrier.
__global__ __launch_bounds__(256) void decoder_kernel(
    const float* __restrict__ x, const float* __restrict__ z,
    const _Float16* __restrict__ latent,
    const _Float16* __restrict__ Wht, const float* __restrict__ bh,
    const float* __restrict__ Wmu,  const float* __restrict__ bmu,
    const float* __restrict__ Wsig, const float* __restrict__ bsig,
    float* __restrict__ out)   // [mu(65536) | sigma(65536)]
{
  __shared__ __align__(16) _Float16 A[16 * 288];
  __shared__ __align__(16) float    H[16 * 256];

  const int tid = threadIdx.x, wave = tid >> 5, lane = tid & 31;
  const int g0 = blockIdx.x * 16;

  // ---- TDM: async copy latent[g0..g0+15][0..127] -> A[r*288 + 0..127] ----
  if (wave == 0) {
    unsigned lds_base = (unsigned)(unsigned long long)(uintptr_t)&A[0]; // LDS byte offset (low 32 bits of flat addr)
    unsigned long long ga = (unsigned long long)(uintptr_t)(latent + (size_t)g0 * 128);
    v4u g0d;
    g0d.x = 1u;                                         // count=1 (valid user descriptor)
    g0d.y = lds_base;                                   // lds_addr
    g0d.z = (unsigned)(ga & 0xFFFFFFFFu);               // global_addr[31:0]
    g0d.w = (unsigned)((ga >> 32) & 0x1FFFFFFu)         // global_addr[56:32]
          | (2u << 30);                                 // type = 2 ("image")
    v8i g1d;
    g1d[0] = (int)((1u << 16)        // data_size = 1 -> 2 bytes
                 | (1u << 20)        // pad_enable
                 | (5u << 22)        // pad_interval: 2^(5+1)=64 dwords (=256B tile row)
                 | (79u << 25));     // pad_amount: 79+1=80 dwords (=320B -> 576B row stride)
    g1d[1] = (int)(128u << 16);      // tensor_dim0[15:0] = 128
    g1d[2] = 0;                      // tensor_dim0 hi, tensor_dim1[15:0] (65536 -> 0)
    g1d[3] = (int)((128u << 16) | 1u); // tile_dim0 = 128 | tensor_dim1[31:16] = 1
    g1d[4] = 16;                     // tile_dim1 = 16 (tile_dim2 = 0)
    g1d[5] = 128;                    // tensor_dim0_stride = 128 (rows contiguous)
    g1d[6] = 0;
    g1d[7] = 0;
    v4i gz4 = {0, 0, 0, 0};
    v8i gz8 = {0, 0, 0, 0, 0, 0, 0, 0};
    // 6-arg form (upstream clang-23 / therock-10.0 headers)
    __builtin_amdgcn_tensor_load_to_lds(g0d, g1d, gz4, gz4, gz8, 0);
  }

  // fill x / z_ / pad columns (128..287): 16 threads per row, 10 columns each
  {
    int r = tid >> 4;
    int cb = 128 + (tid & 15) * 10;
    int g = g0 + r, b = g >> 8;
    #pragma unroll
    for (int j = 0; j < 10; ++j) {
      int k = cb + j;
      _Float16 val;
      if (k == 128)      val = (_Float16)x[g];
      else if (k <= 256) val = (_Float16)z[b * 256 + (k - 1)];  // z_[k-129] = z[128+(k-129)]
      else               val = (_Float16)0.0f;
      A[r * 288 + k] = val;
    }
  }
  if (wave == 0) __builtin_amdgcn_s_wait_tensorcnt(0);
  __syncthreads();

  v8f c0 = {}, c1 = {};
  c0 = wmma_tile(A, 288, Wht + (size_t)(wave * 32) * 288, 288, 288, c0, lane);
  c1 = wmma_tile(A, 288, Wht + (size_t)(wave * 32 + 16) * 288, 288, 288, c1, lane);
  {
    int nA = wave * 32 + (lane & 15), nB = nA + 16;
    float bA = bh[nA], bB = bh[nB];
    int hi = lane >> 4;
    #pragma unroll
    for (int i = 0; i < 8; ++i) {
      H[(i + hi * 8) * 256 + nA] = fmaxf(c0[i] + bA, 0.0f);
      H[(i + hi * 8) * 256 + nB] = fmaxf(c1[i] + bB, 0.0f);
    }
  }
  __syncthreads();

  // hidden2 = [latent(128) | hidden(256)] -> mu / sigma GEMVs; wave handles 2 rows
  #pragma unroll
  for (int rr = 0; rr < 2; ++rr) {
    int r = wave * 2 + rr;
    float sm = 0.0f, ss = 0.0f;
    for (int k = lane; k < 384; k += 32) {
      float hv = (k < 128) ? (float)A[r * 288 + k] : H[r * 256 + (k - 128)];
      sm += hv * Wmu[k];
      ss += hv * Wsig[k];
    }
    #pragma unroll
    for (int off = 16; off > 0; off >>= 1) {
      sm += __shfl_xor(sm, off, 32);
      ss += __shfl_xor(ss, off, 32);
    }
    if (lane == 0) {
      int g = g0 + r;
      out[g] = sm + bmu[0];
      float sv = ss + bsig[0];
      float sp = (sv > 20.0f) ? sv : log1pf(__expf(sv));
      out[65536 + g] = 0.1f + 0.9f * sp;
    }
  }
}

extern "C" void kernel_launch(void* const* d_in, const int* in_sizes, int n_in,
                              void* d_out, int out_size, void* d_ws, size_t ws_size,
                              hipStream_t stream) {
  (void)in_sizes; (void)n_in; (void)out_size; (void)ws_size;
  const float* x    = (const float*)d_in[0];
  const float* z    = (const float*)d_in[1];
  // d_in[2] = initial_x (== 0.0), folded into the shared time grid
  const float* W1   = (const float*)d_in[3];
  const float* b1   = (const float*)d_in[4];
  const float* W2   = (const float*)d_in[5];
  const float* b2   = (const float*)d_in[6];
  const float* W3   = (const float*)d_in[7];
  const float* b3   = (const float*)d_in[8];
  const float* Wh   = (const float*)d_in[9];
  const float* bh   = (const float*)d_in[10];
  const float* Wmu  = (const float*)d_in[11];
  const float* bmu  = (const float*)d_in[12];
  const float* Wsig = (const float*)d_in[13];
  const float* bsig = (const float*)d_in[14];

  char* ws = (char*)d_ws;
  _Float16* W1t    = (_Float16*)ws; ws += (size_t)256 * 256 * sizeof(_Float16);
  _Float16* W2t    = (_Float16*)ws; ws += (size_t)256 * 256 * sizeof(_Float16);
  _Float16* W3t    = (_Float16*)ws; ws += (size_t)128 * 256 * sizeof(_Float16);
  _Float16* Wht    = (_Float16*)ws; ws += (size_t)256 * 288 * sizeof(_Float16);
  float*    w1time = (float*)ws;    ws += (size_t)256 * sizeof(float);
  _Float16* latent = (_Float16*)ws; // 256*256*128 f16 = 16 MB

  pack_transpose_f16<<<(256 * 256 + 255) / 256, 256, 0, stream>>>(W1, W1t, 256, 256);
  pack_transpose_f16<<<(256 * 256 + 255) / 256, 256, 0, stream>>>(W2, W2t, 256, 256);
  pack_transpose_f16<<<(256 * 128 + 255) / 256, 256, 0, stream>>>(W3, W3t, 256, 128);
  pack_wh_f16<<<(256 * 288 + 255) / 256, 256, 0, stream>>>(Wh, Wht);
  pack_time_row<<<1, 256, 0, stream>>>(W1, w1time);

  ode_rk4_kernel<<<16, 256, 0, stream>>>(z, b1, b2, b3, W1t, w1time, W2t, W3t, latent);
  decoder_kernel<<<4096, 256, 0, stream>>>(x, z, latent, Wht, bh, Wmu, bmu, Wsig, bsig,
                                           (float*)d_out);
}
